// InstanceLossSP_51092930953496
// MI455X (gfx1250) — matrix operand order repaired and checked by reading, not compiled, not run on hardware
//
#include <hip/hip_runtime.h>

typedef __attribute__((ext_vector_type(16))) _Float16 v16h;
typedef __attribute__((ext_vector_type(8)))  _Float16 v8h;
typedef __attribute__((ext_vector_type(4)))  _Float16 v4h;
typedef __attribute__((ext_vector_type(8)))  float    v8f;

#define NROWS 12288
#define DDIM  128
#define TEMP_INV 2.0f      // 1 / 0.5
#define NTOP 11            // top-(k+1), k = 10

// ---------------------------------------------------------------------------
// Kernel 1: L2-normalize each row (fp32), emit f16 hi + f16 lo split copies.
// One wave per row: 128 elems = 4 per lane.
// ---------------------------------------------------------------------------
__global__ void norm_split_kernel(const float* __restrict__ z,
                                  _Float16* __restrict__ hi,
                                  _Float16* __restrict__ lo) {
    int wave = (blockIdx.x * blockDim.x + threadIdx.x) >> 5;
    int lane = threadIdx.x & 31;
    if (wave >= NROWS) return;

    const float4 v = ((const float4*)(z + (size_t)wave * DDIM))[lane];
    float ss = v.x * v.x + v.y * v.y + v.z * v.z + v.w * v.w;
    #pragma unroll
    for (int off = 16; off >= 1; off >>= 1)
        ss += __shfl_xor(ss, off, 32);
    float rn = rsqrtf(ss);

    float n0 = v.x * rn, n1 = v.y * rn, n2 = v.z * rn, n3 = v.w * rn;
    v4h h, l;
    h.x = (_Float16)n0; l.x = (_Float16)(n0 - (float)h.x);
    h.y = (_Float16)n1; l.y = (_Float16)(n1 - (float)h.y);
    h.z = (_Float16)n2; l.z = (_Float16)(n2 - (float)h.z);
    h.w = (_Float16)n3; l.w = (_Float16)(n3 - (float)h.w);
    *(v4h*)(hi + (size_t)wave * DDIM + lane * 4) = h;
    *(v4h*)(lo + (size_t)wave * DDIM + lane * 4) = l;
}

// ---------------------------------------------------------------------------
// Kernel 2: fused sim-tile GEMM (WMMA f16 split) + exp + rowsum + top-11.
// Each wave owns a 16-row tile; block = 128 threads = 4 independent waves.
// ---------------------------------------------------------------------------
__global__ void __launch_bounds__(128)
sim_topk_kernel(const _Float16* __restrict__ hi,
                const _Float16* __restrict__ lo,
                float* __restrict__ rowterm) {
    __shared__ float tile[4][16 * 20];   // per-wave C-layout -> row-major bounce
    __shared__ float merge[4][16][24];   // 22 top-k candidates + 2 partial sums

    const int lane  = threadIdx.x & 31;
    const int wslot = threadIdx.x >> 5;
    const int wid   = blockIdx.x * 4 + wslot;   // row-tile id, 0..767
    const int row_base = wid * 16;
    const int lrow = lane & 15;
    const int lgrp = lane >> 4;

    // -------- Load resident A fragments (16 rows x K=128, hi & lo) ----------
    // 16-bit A 16x32 layout: lanes 0-15 halves = K[0..7],K[16..23];
    //                        lanes 16-31 halves = K[8..15],K[24..31].
    v16h Ah[4], Al[4];
    {
        const _Float16* ph = hi + (size_t)(row_base + lrow) * DDIM;
        const _Float16* pl = lo + (size_t)(row_base + lrow) * DDIM;
        #pragma unroll
        for (int kc = 0; kc < 4; ++kc) {
            int k0 = kc * 32 + 8 * lgrp;
            int k1 = kc * 32 + 16 + 8 * lgrp;
            union { v16h v; v8h h[2]; } uh, ul;
            uh.h[0] = *(const v8h*)(ph + k0);
            uh.h[1] = *(const v8h*)(ph + k1);
            ul.h[0] = *(const v8h*)(pl + k0);
            ul.h[1] = *(const v8h*)(pl + k1);
            Ah[kc] = uh.v;
            Al[kc] = ul.v;
        }
    }

    // Per-lane streaming state: top-11 of this lane's column subset + row sum.
    float topv[NTOP];
    #pragma unroll
    for (int i = 0; i < NTOP; ++i) topv[i] = 0.0f;   // sim > 0 always
    float cmin = 0.0f;
    int   cidx = 0;
    float rsum = 0.0f;

    float* my_tile = &tile[wslot][0];

    for (int ct = 0; ct < NROWS / 16; ++ct) {
        const int col_base = ct * 16;
        // ---- B fragments: 16-bit B 32x16 layout: lane = N, lanes 0-15 hold
        //      K[0..7],K[16..23], lanes 16-31 hold K[8..15],K[24..31]. ------
        const _Float16* pbh = hi + (size_t)(col_base + lrow) * DDIM;
        const _Float16* pbl = lo + (size_t)(col_base + lrow) * DDIM;

        v8f C0 = {}, C1 = {}, C2 = {};
        #pragma unroll
        for (int kc = 0; kc < 4; ++kc) {
            int kb = kc * 32 + 8 * lgrp;
            union { v16h v; v8h h[2]; } bh, bl;
            bh.h[0] = *(const v8h*)(pbh + kb);
            bh.h[1] = *(const v8h*)(pbh + kb + 16);
            bl.h[0] = *(const v8h*)(pbl + kb);
            bl.h[1] = *(const v8h*)(pbl + kb + 16);
            // split product: hi*hi + hi*lo + lo*hi  (3 accumulator chains)
            C0 = __builtin_amdgcn_wmma_f32_16x16x32_f16(false, Ah[kc], false, bh.v,
                                                        (short)0, C0, false, false);
            C1 = __builtin_amdgcn_wmma_f32_16x16x32_f16(false, Ah[kc], false, bl.v,
                                                        (short)0, C1, false, false);
            C2 = __builtin_amdgcn_wmma_f32_16x16x32_f16(false, Al[kc], false, bh.v,
                                                        (short)0, C2, false, false);
        }

        // ---- exp(2*dot), bounce through LDS to get row-major per-lane ------
        // C layout: lane holds col=lrow, rows r+8*lgrp for r=0..7.
        #pragma unroll
        for (int r = 0; r < 8; ++r) {
            float d = C0[r] + C1[r] + C2[r];
            float e = __expf(d * TEMP_INV);
            my_tile[(r + 8 * lgrp) * 20 + lrow] = e;
        }
        __builtin_amdgcn_wave_barrier();   // keep LDS store->load ordered
        float4 v0 = *(const float4*)(my_tile + lrow * 20 + 8 * lgrp);
        float4 v1 = *(const float4*)(my_tile + lrow * 20 + 8 * lgrp + 4);
        __builtin_amdgcn_wave_barrier();   // don't let next stores pass loads
        float vv[8] = {v0.x, v0.y, v0.z, v0.w, v1.x, v1.y, v1.z, v1.w};

        #pragma unroll
        for (int i = 0; i < 8; ++i) {
            float e = vv[i];
            rsum += e;
            if (e > cmin) {
                // replace current min (constant-index select chain: stays in VGPRs)
                #pragma unroll
                for (int j = 0; j < NTOP; ++j)
                    topv[j] = (j == cidx) ? e : topv[j];
                cmin = topv[0]; cidx = 0;
                #pragma unroll
                for (int j = 1; j < NTOP; ++j)
                    if (topv[j] < cmin) { cmin = topv[j]; cidx = j; }
            }
        }
    }

    // -------- finalize: merge the two half-column lists per row -------------
    float (*mrg)[24] = merge[wslot];
    #pragma unroll
    for (int i = 0; i < NTOP; ++i) mrg[lrow][lgrp * NTOP + i] = topv[i];
    mrg[lrow][22 + lgrp] = rsum;
    __builtin_amdgcn_wave_barrier();

    if (lane < 16) {
        float cand[22];
        #pragma unroll
        for (int i = 0; i < 22; ++i) cand[i] = mrg[lane][i];
        float total = mrg[lane][22] + mrg[lane][23];

        float s11 = 0.0f, mx = 0.0f;
        for (int t = 0; t < NTOP; ++t) {
            float bv = cand[0]; int bi = 0;
            #pragma unroll
            for (int j = 1; j < 22; ++j)
                if (cand[j] > bv) { bv = cand[j]; bi = j; }
            s11 += bv;
            mx = (t == 0) ? bv : mx;
            #pragma unroll
            for (int j = 0; j < 22; ++j)
                cand[j] = (j == bi) ? -1.0f : cand[j];
        }
        // mx == sim[i,i] (self is the strict row max): drop it from both sums.
        float e_sim = s11 - mx;
        float e_all = total - mx;
        rowterm[row_base + lane] = __logf(e_all) - __logf(e_sim);
    }
}

// ---------------------------------------------------------------------------
// Kernel 3: deterministic mean of the 12288 per-row terms (single block).
// ---------------------------------------------------------------------------
__global__ void reduce_mean_kernel(const float* __restrict__ terms,
                                   float* __restrict__ out) {
    __shared__ float sm[256];
    float s = 0.0f;
    for (int i = threadIdx.x; i < NROWS; i += 256) s += terms[i];
    sm[threadIdx.x] = s;
    __syncthreads();
    for (int off = 128; off >= 1; off >>= 1) {
        if ((int)threadIdx.x < off) sm[threadIdx.x] += sm[threadIdx.x + off];
        __syncthreads();
    }
    if (threadIdx.x == 0) out[0] = sm[0] / (float)NROWS;
}

// ---------------------------------------------------------------------------
extern "C" void kernel_launch(void* const* d_in, const int* in_sizes, int n_in,
                              void* d_out, int out_size, void* d_ws, size_t ws_size,
                              hipStream_t stream) {
    (void)in_sizes; (void)n_in; (void)out_size; (void)ws_size;
    const float* z = (const float*)d_in[0];
    float* out = (float*)d_out;

    char* ws = (char*)d_ws;
    _Float16* hi    = (_Float16*)ws;                                   // 3 MB
    _Float16* lo    = (_Float16*)(ws + (size_t)NROWS * DDIM * 2);      // 3 MB
    float*    terms = (float*)   (ws + (size_t)NROWS * DDIM * 4);      // 48 KB

    norm_split_kernel<<<NROWS / 8, 256, 0, stream>>>(z, hi, lo);
    sim_topk_kernel<<<(NROWS / 16) / 4, 128, 0, stream>>>(hi, lo, terms);
    reduce_mean_kernel<<<1, 256, 0, stream>>>(terms, out);
}
